// ContractiveEquivariantMPlayer_68272800137474
// MI455X (gfx1250) — matrix operand order repaired and checked by reading, not compiled, not run on hardware
//
#include <hip/hip_runtime.h>
#include <hip/hip_bf16.h>

typedef __attribute__((ext_vector_type(16))) _Float16 v16h;
typedef __attribute__((ext_vector_type(8)))  _Float16 v8h;
typedef __attribute__((ext_vector_type(8)))  float    v8f;

namespace ceq {

constexpr int FD = 128;      // feature dim
constexpr int GR = 50;       // n_rbf
constexpr float CUT = 5.0f;  // cutoff

constexpr int WSTR1 = 72;    // LDS stride (f16) of transposed Wf1 (K=64 + 8 pad)
constexpr int WSTR  = 136;   // LDS stride (f16) of transposed 128x128 weights
constexpr int SSTR  = 136;   // LDS stride (f16) of the 16-row stage tile

__device__ __forceinline__ v8f wmma16(v16h a, v16h b, v8f c) {
  return __builtin_amdgcn_wmma_f32_16x16x32_f16(
      false, a, false, b, (short)0, c, false, false);
}

__device__ __forceinline__ v16h cat8(v8h lo, v8h hi) {
  return __builtin_shufflevector(lo, hi, 0, 1, 2, 3, 4, 5, 6, 7,
                                 8, 9, 10, 11, 12, 13, 14, 15);
}

// A-fragment (16x32 f16 MxK) from row-major f16 stage tile (stride SSTR).
// Lane l: row M=l&15; values are two contiguous 16B runs:
//   K = kbase+8*half .. +7   and   K = kbase+16+8*half .. +7
__device__ __forceinline__ v16h load_a_frag(const _Float16* __restrict__ t,
                                            int kbase, int lane) {
  const int M = lane & 15;
  const int half = lane >> 4;
  const _Float16* p = t + M * SSTR + kbase + (half << 3);
  const v8h lo = *(const v8h*)(p);       // ds_load_b128
  const v8h hi = *(const v8h*)(p + 16);  // ds_load_b128
  return cat8(lo, hi);
}

// B-fragment (32x16 f16 KxN) from TRANSPOSED (K-contiguous) f16 weights.
// Lane l: col N=nbase+(l&15); values are two contiguous 16B runs:
//   K = kbase+8*half .. +7   and   K = kbase+16+8*half .. +7
__device__ __forceinline__ v16h load_b_fragT(const _Float16* __restrict__ wT,
                                             int stride, int kbase, int nbase,
                                             int lane) {
  const int N = nbase + (lane & 15);
  const int half = lane >> 4;
  const _Float16* p = wT + N * stride + kbase + (half << 3);
  const v8h lo = *(const v8h*)(p);      // ds_load_b128
  const v8h hi = *(const v8h*)(p + 16);  // ds_load_b128
  return cat8(lo, hi);
}

// Gaussian-RBF A-fragment computed directly in registers (K>=50 -> 0).
__device__ __forceinline__ v16h rbf_a_frag(float d, int kbase, int lane) {
  const int half = lane >> 4;
  const float width = CUT / (float)(GR - 1);
  const float coeff = -0.5f / (width * width);
  v16h a;
#pragma unroll
  for (int j = 0; j < 8; ++j) {
    const int kk = kbase + ((j >> 2) << 4) + (half << 3) + ((j & 3) << 1);
#pragma unroll
    for (int e = 0; e < 2; ++e) {
      const int K = kk + e;
      const float t = d - (float)K * width;
      const float v = (K < GR) ? __expf(coeff * t * t) : 0.0f;
      a[2 * j + e] = (_Float16)v;
    }
  }
  return a;
}

// Branchless, overflow-safe shifted softplus: softplus(x) - ln2
//   = max(x,0) + ln2*(log2(1 + 2^(-|x|*log2e)) - 1)
// Uses raw v_exp_f32 / v_log_f32 (no divergent range-fixup code).
__device__ __forceinline__ float shifted_softplus(float x) {
  const float LOG2E = 1.4426950408889634f;
  const float LN2 = 0.6931471805599453f;
  const float e = __builtin_amdgcn_exp2f(-fabsf(x) * LOG2E);
  return fmaxf(x, 0.0f) + LN2 * (__builtin_amdgcn_logf(1.0f + e) - 1.0f);
}

__device__ __forceinline__ v8f bcast8(float x) {
  v8f c = {x, x, x, x, x, x, x, x};
  return c;
}

// ---------------------------------------------------------------------------
__global__ void zero_kernel(float* __restrict__ out, int n_out,
                            float* __restrict__ cnt, int n_beads) {
  const int i = blockIdx.x * blockDim.x + threadIdx.x;
  if (i < n_out) out[i] = 0.0f;
  if (i < n_beads) cnt[i] = 0.0f;
}

__global__ void count_kernel(const int* __restrict__ mapping,
                             float* __restrict__ cnt, int n_atoms) {
  const int i = blockIdx.x * blockDim.x + threadIdx.x;
  if (i < n_atoms) atomicAdd(&cnt[mapping[i]], 1.0f);
}

// ---------------------------------------------------------------------------
// Main WMMA pipeline. One wave32 per 16-atom tile, 8 waves per block.
__global__ __launch_bounds__(256) void edge_filter_kernel(
    const float* __restrict__ h,    const float* __restrict__ v,
    const float* __restrict__ dist, const float* __restrict__ ur,
    const float* __restrict__ Wf1,  const float* __restrict__ bf1,
    const float* __restrict__ Wf2,  const float* __restrict__ bf2,
    const float* __restrict__ Wa,   const float* __restrict__ ba,
    const float* __restrict__ Wb,   const float* __restrict__ bb,
    const int*   __restrict__ mapping,
    float* __restrict__ out_dh, float* __restrict__ out_dv,
    int n_atoms, int n_tiles) {
  // Transposed (K-contiguous) f16 weights in LDS, padded strides for
  // conflict-free ds_load_b128 B-fragment reads.
  __shared__ __align__(16) _Float16 sWf1T[FD * WSTR1];  // 18 KB
  __shared__ __align__(16) _Float16 sWf2T[FD * WSTR];   // 34 KB
  __shared__ __align__(16) _Float16 sWaT[FD * WSTR];    // 34 KB
  __shared__ __align__(16) _Float16 sWbT[FD * WSTR];    // 34 KB
  __shared__ __align__(16) _Float16 stage[8][16 * SSTR];  // 4.25 KB / wave

  // Cooperative transpose + f32->f16 conversion (coalesced global reads).
  for (int i = threadIdx.x; i < FD * 64; i += 256) {
    const int k = i >> 7, n = i & (FD - 1);
    sWf1T[n * WSTR1 + k] = (_Float16)(k < GR ? Wf1[k * FD + n] : 0.0f);
  }
  for (int i = threadIdx.x; i < FD * FD; i += 256) {
    const int k = i >> 7, n = i & (FD - 1);
    sWf2T[n * WSTR + k] = (_Float16)Wf2[i];
    sWaT[n * WSTR + k]  = (_Float16)Wa[i];
    sWbT[n * WSTR + k]  = (_Float16)Wb[i];
  }
  __syncthreads();

  const int wave = threadIdx.x >> 5;
  const int lane = threadIdx.x & 31;
  const int tile = blockIdx.x * 8 + wave;
  if (tile >= n_tiles) return;  // wave-uniform: EXEC all-ones for WMMA

  const int base = tile * 16;
  const int M = lane & 15;
  const int half = lane >> 4;
  _Float16* S = &stage[wave][0];

  // --- RBF A-fragments straight from distances (K padded 50->64) ---
  const float dval = dist[min(base + M, n_atoms - 1)];
  const v16h ra0 = rbf_a_frag(dval, 0, lane);
  const v16h ra1 = rbf_a_frag(dval, 32, lane);

  // --- GEMM1: t1 = ssp(rbf @ Wf1 + bf1) -> stage tile (f16) ---
#pragma unroll
  for (int n = 0; n < 8; ++n) {
    v8f c = bcast8(bf1[n * 16 + M]);
    c = wmma16(ra0, load_b_fragT(sWf1T, WSTR1, 0, n * 16, lane), c);
    c = wmma16(ra1, load_b_fragT(sWf1T, WSTR1, 32, n * 16, lane), c);
#pragma unroll
    for (int j = 0; j < 8; ++j) {  // C layout: row j+8*half, col n*16+M
      S[(j + 8 * half) * SSTR + n * 16 + M] = (_Float16)shifted_softplus(c[j]);
    }
  }
  asm volatile("s_wait_dscnt 0x0" ::: "memory");  // wave-order DS fence

  // --- GEMM2: Wfilt = t1 @ Wf2 + bf2 ; edge_inv = h * Wfilt -> stage ---
  v16h a0 = load_a_frag(S, 0, lane);
  v16h a1 = load_a_frag(S, 32, lane);
  v16h a2 = load_a_frag(S, 64, lane);
  v16h a3 = load_a_frag(S, 96, lane);  // A in regs; S safely overwritten below
#pragma unroll
  for (int n = 0; n < 8; ++n) {
    v8f c = bcast8(bf2[n * 16 + M]);
    c = wmma16(a0, load_b_fragT(sWf2T, WSTR, 0, n * 16, lane), c);
    c = wmma16(a1, load_b_fragT(sWf2T, WSTR, 32, n * 16, lane), c);
    c = wmma16(a2, load_b_fragT(sWf2T, WSTR, 64, n * 16, lane), c);
    c = wmma16(a3, load_b_fragT(sWf2T, WSTR, 96, n * 16, lane), c);
#pragma unroll
    for (int j = 0; j < 8; ++j) {
      const int atom = min(base + j + 8 * half, n_atoms - 1);
      const float hv = h[atom * FD + n * 16 + M];
      S[(j + 8 * half) * SSTR + n * 16 + M] = (_Float16)(hv * c[j]);
    }
  }
  asm volatile("s_wait_dscnt 0x0" ::: "memory");

  // --- GEMM3/4: fa = ei@Wa + ba ; fb = ei@Wb + bb ; fused atomic scatter ---
  a0 = load_a_frag(S, 0, lane);
  a1 = load_a_frag(S, 32, lane);
  a2 = load_a_frag(S, 64, lane);
  a3 = load_a_frag(S, 96, lane);

  // Hoist j-only scatter operands (invariant across the 8 n-tiles).
  int beadj[8];
  float u0j[8], u1j[8], u2j[8];
#pragma unroll
  for (int j = 0; j < 8; ++j) {
    const int atom = min(base + j + 8 * half, n_atoms - 1);
    beadj[j] = mapping[atom];
    u0j[j] = ur[atom * 3 + 0];
    u1j[j] = ur[atom * 3 + 1];
    u2j[j] = ur[atom * 3 + 2];
  }

#pragma unroll
  for (int n = 0; n < 8; ++n) {
    v8f ca = bcast8(ba[n * 16 + M]);
    v8f cb = bcast8(bb[n * 16 + M]);
    ca = wmma16(a0, load_b_fragT(sWaT, WSTR, 0, n * 16, lane), ca);
    cb = wmma16(a0, load_b_fragT(sWbT, WSTR, 0, n * 16, lane), cb);
    ca = wmma16(a1, load_b_fragT(sWaT, WSTR, 32, n * 16, lane), ca);
    cb = wmma16(a1, load_b_fragT(sWbT, WSTR, 32, n * 16, lane), cb);
    ca = wmma16(a2, load_b_fragT(sWaT, WSTR, 64, n * 16, lane), ca);
    cb = wmma16(a2, load_b_fragT(sWbT, WSTR, 64, n * 16, lane), cb);
    ca = wmma16(a3, load_b_fragT(sWaT, WSTR, 96, n * 16, lane), ca);
    cb = wmma16(a3, load_b_fragT(sWbT, WSTR, 96, n * 16, lane), cb);
#pragma unroll
    for (int j = 0; j < 8; ++j) {
      const int atom = base + j + 8 * half;
      if (atom >= n_atoms) continue;
      const int feat = n * 16 + M;
      const float fav = ca[j];
      const float fbv = cb[j];
      const float* vv = &v[(size_t)(atom * FD + feat) * 3];  // global_load_b96
      const float v0 = vv[0], v1 = vv[1], v2 = vv[2];
      atomicAdd(&out_dh[(size_t)beadj[j] * FD + feat], fbv);
      float* od = &out_dv[(size_t)(beadj[j] * FD + feat) * 3];
      atomicAdd(&od[0], fav * u0j[j] + fbv * v0);
      atomicAdd(&od[1], fav * u1j[j] + fbv * v1);
      atomicAdd(&od[2], fav * u2j[j] + fbv * v2);
    }
  }
}

// ---------------------------------------------------------------------------
__global__ void finalize_kernel(float* __restrict__ dh, float* __restrict__ dv,
                                const float* __restrict__ cnt, int n_beads) {
  const int i = blockIdx.x * blockDim.x + threadIdx.x;
  if (i >= n_beads * FD) return;
  const float s = 1.0f / fmaxf(cnt[i >> 7], 1.0f);
  dh[i] *= s;
  dv[3 * i + 0] *= s;
  dv[3 * i + 1] *= s;
  dv[3 * i + 2] *= s;
}

}  // namespace ceq

extern "C" void kernel_launch(void* const* d_in, const int* in_sizes, int n_in,
                              void* d_out, int out_size, void* d_ws, size_t ws_size,
                              hipStream_t stream) {
  using namespace ceq;
  const float* h    = (const float*)d_in[0];
  const float* v    = (const float*)d_in[1];
  const float* dist = (const float*)d_in[2];
  const float* ur   = (const float*)d_in[3];
  const float* Wf1  = (const float*)d_in[4];
  const float* bf1  = (const float*)d_in[5];
  const float* Wf2  = (const float*)d_in[6];
  const float* bf2  = (const float*)d_in[7];
  // d_in[8..11] = W1,b1,W2,b2 : dead code in reference (phi unused) -> skipped
  const float* Wa   = (const float*)d_in[12];
  const float* ba   = (const float*)d_in[13];
  const float* Wb   = (const float*)d_in[14];
  const float* bb   = (const float*)d_in[15];
  const int* mapping = (const int*)d_in[16];

  const int n_atoms = in_sizes[0] / FD;     // 200000
  const int n_beads = out_size / (FD * 4);  // dh: B*F, dv: B*F*3
  float* out_dh = (float*)d_out;
  float* out_dv = out_dh + (size_t)n_beads * FD;
  float* cnt = (float*)d_ws;

  {
    const int n = out_size > n_beads ? out_size : n_beads;
    zero_kernel<<<(n + 255) / 256, 256, 0, stream>>>(out_dh, out_size, cnt, n_beads);
  }
  count_kernel<<<(n_atoms + 255) / 256, 256, 0, stream>>>(mapping, cnt, n_atoms);
  {
    const int n_tiles = (n_atoms + 15) / 16;
    const int blocks = (n_tiles + 7) / 8;
    edge_filter_kernel<<<blocks, 256, 0, stream>>>(
        h, v, dist, ur, Wf1, bf1, Wf2, bf2, Wa, ba, Wb, bb,
        mapping, out_dh, out_dv, n_atoms, n_tiles);
  }
  {
    const int n = n_beads * FD;
    finalize_kernel<<<(n + 255) / 256, 256, 0, stream>>>(out_dh, out_dv, cnt, n_beads);
  }
}
